// Gemma4DecoderLayer_12326556139559
// MI455X (gfx1250) — matrix operand order repaired and checked by reading, not compiled, not run on hardware
//
#include <hip/hip_runtime.h>

// Problem constants (from reference)
#define T_TOK 2048
#define HID   2048
#define NH    16
#define NKV   8
#define HD    256
#define QKV_W ((NH + 2*NKV)*HD)   // 8192
#define OD    (NH*HD)             // 4096

typedef __bf16 bf16;
typedef __attribute__((ext_vector_type(16))) __bf16 v16bf;
typedef __attribute__((ext_vector_type(8)))  __bf16 v8bf;
typedef __attribute__((ext_vector_type(8)))  float  v8f;

__device__ __forceinline__ v16bf cat8(v8bf lo, v8bf hi) {
    return __builtin_shufflevector(lo, hi, 0,1,2,3,4,5,6,7,8,9,10,11,12,13,14,15);
}
__device__ __forceinline__ v8f zero8() {
    v8f z;
#pragma unroll
    for (int e = 0; e < 8; ++e) z[e] = 0.0f;
    return z;
}

// CDNA5 async DMA: global memory -> LDS, 16B per lane, tracked by ASYNCcnt.
// lds_off is the wave-relative LDS byte offset (low 32 bits of the generic
// pointer per ISA aperture rules), gaddr the 64-bit global VA (GV mode).
__device__ __forceinline__ void async_load_b128(unsigned lds_off, const void* gptr) {
    asm volatile("global_load_async_to_lds_b128 %0, %1, off"
                 :: "v"(lds_off), "v"((unsigned long long)(uintptr_t)gptr)
                 : "memory");
}
__device__ __forceinline__ void wait_async0() {
    asm volatile("s_wait_asynccnt 0x0" ::: "memory");
}

// ---------------------------------------------------------------------------
// fp32 -> bf16 conversion (grid-stride)
// ---------------------------------------------------------------------------
__global__ void cvt_f32_bf16(const float* __restrict__ in, bf16* __restrict__ out, int n) {
    int i = blockIdx.x * blockDim.x + threadIdx.x;
    const int stride = gridDim.x * blockDim.x;
    for (; i < n; i += stride) out[i] = (bf16)in[i];
}

// ---------------------------------------------------------------------------
// Tiled bf16 WMMA GEMM: C[M,N] (f32) = A[M,K] x B[K,N], both bf16 row-major.
// Block tile 128x128x32, 256 threads = 8 waves; wave w computes rows
// [w*16, w*16+16) x 128 cols => 8 accumulator tiles of 16x16.
// Staging: double-buffered LDS filled by GLOBAL_LOAD_ASYNC_TO_LDS_B128 (DMA),
// one s_wait_asynccnt + one barrier per K-step. sched_group_barrier forces
// 18 of the 130 ds_loads ahead of the 8 WMMAs so dscnt waits become progressive.
// ---------------------------------------------------------------------------
__global__ __launch_bounds__(256)
void gemm_bf16_kernel(const bf16* __restrict__ A, const bf16* __restrict__ B,
                      float* __restrict__ C, int M, int N, int K)
{
    __shared__ __align__(16) bf16 As[2][128][40];   // 32 cols + pad
    __shared__ __align__(16) bf16 Bs[2][32][136];   // 128 cols + pad

    const int tid   = threadIdx.x;
    const int lane  = tid & 31;
    const int w     = tid >> 5;
    const int m0    = blockIdx.y * 128;
    const int n0    = blockIdx.x * 128;
    const int lrow  = lane & 15;
    const int lhalf = lane >> 4;

    v8f acc[8];
#pragma unroll
    for (int j = 0; j < 8; ++j) acc[j] = zero8();

    const int aRow = tid >> 1, aCol = (tid & 1) * 16;   // 128x32 tile, 16 elem/thread
    const int bRow = tid >> 3, bCol = (tid & 7) * 16;   // 32x128 tile, 16 elem/thread

    auto stage = [&](int buf, int k0) {
        const bf16* ag = &A[(size_t)(m0 + aRow) * K + k0 + aCol];
        const bf16* bg = &B[(size_t)(k0 + bRow) * N + n0 + bCol];
        const unsigned al = (unsigned)(uintptr_t)&As[buf][aRow][aCol];
        const unsigned bl = (unsigned)(uintptr_t)&Bs[buf][bRow][bCol];
        async_load_b128(al,      ag);
        async_load_b128(al + 16, ag + 8);
        async_load_b128(bl,      bg);
        async_load_b128(bl + 16, bg + 8);
    };

    stage(0, 0);
    int buf = 0;
    for (int k0 = 0; k0 < K; k0 += 32) {
        wait_async0();          // own wave's DMA into As[buf]/Bs[buf] done
        __syncthreads();        // everyone's DMA done, previous readers done
        if (k0 + 32 < K) stage(buf ^ 1, k0 + 32);   // overlap DMA with math

        // A fragment (16x32): lane layout per ISA 7.12.2
        v8bf alo = *(const v8bf*)&As[buf][w * 16 + lrow][lhalf * 8];
        v8bf ahi = *(const v8bf*)&As[buf][w * 16 + lrow][16 + lhalf * 8];
        v16bf af = cat8(alo, ahi);

        // Pull all 8 B fragments into registers, then chain the WMMAs.
        v16bf bfr[8];
#pragma unroll
        for (int j = 0; j < 8; ++j) {
#pragma unroll
            for (int s = 0; s < 16; ++s) {
                const int kk = (s < 8) ? lhalf * 8 + s : 16 + lhalf * 8 + (s - 8);
                bfr[j][s] = Bs[buf][kk][j * 16 + lrow];
            }
        }
#pragma unroll
        for (int j = 0; j < 8; ++j)
            acc[j] = __builtin_amdgcn_wmma_f32_16x16x32_bf16(
                false, af, false, bfr[j], (short)0, acc[j], false, false);

        // Scheduling directive: issue 18 DS reads first (of 130:
        // 2 for A frag + 128 for B frags), then the 8 WMMAs, so the waits
        // become progressive instead of dscnt==0 before each matrix op.
        __builtin_amdgcn_sched_group_barrier(0x100, 18, 0);  // DS reads
        __builtin_amdgcn_sched_group_barrier(0x008, 8, 0);   // WMMA

        buf ^= 1;
    }

#pragma unroll
    for (int j = 0; j < 8; ++j)
#pragma unroll
        for (int r = 0; r < 8; ++r) {
            const int row = m0 + w * 16 + r + ((lane >= 16) ? 8 : 0);
            const int col = n0 + j * 16 + lrow;
            C[(size_t)row * N + col] = acc[j][r];
        }
}

// ---------------------------------------------------------------------------
// Fused RMSNorm + NeoX RoPE. One block per (token, head-slot).
// Head slots: 0..15 = Q, 16..23 = K, 24..31 = V.
// Q out:  [H][T][D]  bf16
// K out:  [KV][D][T] bf16 (transposed for QK^T B-fragments)
// V out:  [KV][T][D] bf16
// ---------------------------------------------------------------------------
__global__ __launch_bounds__(256)
void norm_rope_kernel(const float* __restrict__ qkv, const int* __restrict__ positions,
                      const float* __restrict__ qw, const float* __restrict__ kw,
                      bf16* __restrict__ qout, bf16* __restrict__ ktout,
                      bf16* __restrict__ vout)
{
    const int t  = blockIdx.x;
    const int hh = blockIdx.y;       // 0..31
    const int i  = threadIdx.x;      // 0..255

    __shared__ float red[256];
    __shared__ float xs[256];

    const float x = qkv[(size_t)t * QKV_W + hh * HD + i];
    red[i] = x * x;
    __syncthreads();
#pragma unroll
    for (int s = 128; s > 0; s >>= 1) {
        if (i < s) red[i] += red[i + s];
        __syncthreads();
    }
    const float inv = rsqrtf(red[0] / (float)HD + 1e-6f);

    float wgt = 1.0f;
    if (hh < NH) wgt = qw[i];
    else if (hh < NH + NKV) wgt = kw[i];
    const float xn = x * inv * wgt;

    if (hh >= NH + NKV) {             // V: rmsnorm only (uniform branch per block)
        const int kv = hh - NH - NKV;
        vout[((size_t)kv * T_TOK + t) * HD + i] = (bf16)xn;
        return;
    }

    xs[i] = xn;
    __syncthreads();

    if (i < HD / 2) {
        const float pos  = (float)positions[t];
        const float freq = __expf(-(2.0f * (float)i / (float)HD) * __logf(10000.0f));
        float s, c;
        __sincosf(pos * freq, &s, &c);
        const float x1 = xs[i], x2 = xs[i + HD / 2];
        const float o1 = x1 * c - x2 * s;
        const float o2 = x2 * c + x1 * s;
        if (hh < NH) {
            bf16* dst = qout + ((size_t)hh * T_TOK + t) * HD;
            dst[i]          = (bf16)o1;
            dst[i + HD / 2] = (bf16)o2;
        } else {
            const int kv = hh - NH;
            ktout[((size_t)kv * HD + i) * T_TOK + t]          = (bf16)o1;
            ktout[((size_t)kv * HD + i + HD / 2) * T_TOK + t] = (bf16)o2;
        }
    }
}

// ---------------------------------------------------------------------------
// Causal flash attention. grid = (T/64, H), 128 threads (4 waves).
// Wave w owns queries [blockIdx.x*64 + w*16, +16). Keys in blocks of 32.
// QK^T: 16 WMMAs/key-block; PV: 16 WMMAs/key-block. Online softmax staged in
// LDS (also used to transpose the C-layout score tile into an A-fragment).
// ---------------------------------------------------------------------------
__global__ __launch_bounds__(128)
void attn_kernel(const bf16* __restrict__ q, const bf16* __restrict__ kt,
                 const bf16* __restrict__ v, bf16* __restrict__ aout)
{
    const int h     = blockIdx.y;
    const int kvh   = h >> 1;               // H/KV = 2
    const int w     = threadIdx.x >> 5;
    const int lane  = threadIdx.x & 31;
    const int lrow  = lane & 15;
    const int lhalf = lane >> 4;
    const int qbase = blockIdx.x * 64;
    const int q0    = qbase + w * 16;
    const float NEG_INF = -__builtin_inff();

    __shared__ float Sld[4][16][33];
    __shared__ __align__(16) bf16 Pld[4][16][40];
    __shared__ float auxScale[4][16];
    __shared__ float auxL[4][16];

    // Q fragments for all 8 K-chunks of D=256, kept in registers (reused).
    v16bf qf[8];
    {
        const bf16* qp = q + ((size_t)h * T_TOK + (q0 + lrow)) * HD;
#pragma unroll
        for (int dk = 0; dk < 8; ++dk) {
            v8bf lo = *(const v8bf*)(qp + dk * 32 + lhalf * 8);
            v8bf hi = *(const v8bf*)(qp + dk * 32 + 16 + lhalf * 8);
            qf[dk] = cat8(lo, hi);
        }
    }

    v8f acc[16];                       // 16 D-tiles of 16x16 => 128 VGPRs fp32
#pragma unroll
    for (int j = 0; j < 16; ++j) acc[j] = zero8();

    float m_row = NEG_INF, l_row = 0.0f;
    const int kend = qbase + 64;       // uniform trip count across the block

    for (int kb = 0; kb < kend; kb += 32) {
        // ---- S = Q K^T for 16 queries x 32 keys ----
#pragma unroll
        for (int sub = 0; sub < 2; ++sub) {
            v8f s = zero8();
            const int n0 = kb + sub * 16;
#pragma unroll
            for (int dk = 0; dk < 8; ++dk) {
                v16bf kf;
#pragma unroll
                for (int e = 0; e < 16; ++e) {
                    const int kk = (e < 8) ? lhalf * 8 + e : 16 + lhalf * 8 + (e - 8);
                    kf[e] = kt[((size_t)kvh * HD + dk * 32 + kk) * T_TOK + n0 + lrow];
                }
                s = __builtin_amdgcn_wmma_f32_16x16x32_bf16(
                    false, qf[dk], false, kf, (short)0, s, false, false);
            }
            const int key = n0 + lrow;
#pragma unroll
            for (int r = 0; r < 8; ++r) {
                const int row = (lane < 16) ? r : r + 8;
                Sld[w][row][sub * 16 + lrow] = (key <= q0 + row) ? s[r] : NEG_INF;
            }
        }
        __syncthreads();

        // ---- online softmax (one lane per row) ----
        if (lane < 16) {
            float mx = NEG_INF;
#pragma unroll
            for (int c = 0; c < 32; ++c) mx = fmaxf(mx, Sld[w][lrow][c]);
            const float mnew = fmaxf(m_row, mx);
            const float mref = (mnew == NEG_INF) ? 0.0f : mnew;
            const float sc   = __expf(m_row - mref);
            float lsum = 0.0f;
#pragma unroll
            for (int c = 0; c < 32; ++c) {
                const float p = __expf(Sld[w][lrow][c] - mref);
                Pld[w][lrow][c] = (bf16)p;
                lsum += p;
            }
            l_row = l_row * sc + lsum;
            m_row = mnew;
            auxScale[w][lrow] = sc;
        }
        __syncthreads();

        // ---- rescale accumulators, then acc += P * V ----
        float rsc[8];
#pragma unroll
        for (int r = 0; r < 8; ++r) rsc[r] = auxScale[w][(lane < 16) ? r : r + 8];
#pragma unroll
        for (int j = 0; j < 16; ++j)
#pragma unroll
            for (int r = 0; r < 8; ++r) acc[j][r] *= rsc[r];

        v8bf plo = *(const v8bf*)&Pld[w][lrow][lhalf * 8];
        v8bf phi = *(const v8bf*)&Pld[w][lrow][16 + lhalf * 8];
        v16bf pf = cat8(plo, phi);
#pragma unroll
        for (int j = 0; j < 16; ++j) {
            v16bf vf;
#pragma unroll
            for (int e = 0; e < 16; ++e) {
                const int kk = (e < 8) ? lhalf * 8 + e : 16 + lhalf * 8 + (e - 8);
                vf[e] = v[((size_t)kvh * T_TOK + kb + kk) * HD + j * 16 + lrow];
            }
            acc[j] = __builtin_amdgcn_wmma_f32_16x16x32_bf16(
                false, pf, false, vf, (short)0, acc[j], false, false);
        }
        __syncthreads();   // LDS reused next iteration
    }

    if (lane < 16) auxL[w][lrow] = l_row;
    __syncthreads();

    float rinv[8];
#pragma unroll
    for (int r = 0; r < 8; ++r) rinv[r] = 1.0f / auxL[w][(lane < 16) ? r : r + 8];
#pragma unroll
    for (int j = 0; j < 16; ++j)
#pragma unroll
        for (int r = 0; r < 8; ++r) {
            const int row = q0 + ((lane < 16) ? r : r + 8);
            const int col = j * 16 + lrow;
            aout[(size_t)row * OD + h * HD + col] = (bf16)(acc[j][r] * rinv[r]);
        }
}

// ---------------------------------------------------------------------------
// Launch
// ---------------------------------------------------------------------------
extern "C" void kernel_launch(void* const* d_in, const int* in_sizes, int n_in,
                              void* d_out, int out_size, void* d_ws, size_t ws_size,
                              hipStream_t stream)
{
    const float* hidden    = (const float*)d_in[0];
    const int*   positions = (const int*)d_in[1];
    const float* w_qkv     = (const float*)d_in[2];
    const float* w_o       = (const float*)d_in[3];
    const float* qn        = (const float*)d_in[4];
    const float* kn        = (const float*)d_in[5];
    float* out = (float*)d_out;
    char*  ws  = (char*)d_ws;

    const size_t MB = (size_t)1 << 20;
    bf16*  hid_bf  = (bf16*)(ws);              //   8 MB  [T][HID]
    bf16*  wqkv_bf = (bf16*)(ws + 8  * MB);    //  32 MB  [HID][8192]
    bf16*  wo_bf   = (bf16*)(ws + 40 * MB);    //  16 MB  [4096][HID]
    float* qkv     = (float*)(ws + 56 * MB);   //  64 MB  [T][8192]
    bf16*  attn_bf = (bf16*)(ws + 56 * MB);    //  16 MB, aliases qkv (dead by then)
    bf16*  q_bf    = (bf16*)(ws + 120 * MB);   //  16 MB  [H][T][D]
    bf16*  kt_bf   = (bf16*)(ws + 136 * MB);   //   8 MB  [KV][D][T]
    bf16*  v_bf    = (bf16*)(ws + 144 * MB);   //   8 MB  [KV][T][D]
    // peak workspace: 152 MB

    cvt_f32_bf16<<<2048, 256, 0, stream>>>(hidden, hid_bf, T_TOK * HID);
    cvt_f32_bf16<<<4096, 256, 0, stream>>>(w_qkv, wqkv_bf, HID * QKV_W);
    cvt_f32_bf16<<<4096, 256, 0, stream>>>(w_o, wo_bf, OD * HID);

    gemm_bf16_kernel<<<dim3(QKV_W / 128, T_TOK / 128), 256, 0, stream>>>(
        hid_bf, wqkv_bf, qkv, T_TOK, QKV_W, HID);

    norm_rope_kernel<<<dim3(T_TOK, NH + 2 * NKV), 256, 0, stream>>>(
        qkv, positions, qn, kn, q_bf, kt_bf, v_bf);

    attn_kernel<<<dim3(T_TOK / 64, NH), 128, 0, stream>>>(q_bf, kt_bf, v_bf, attn_bf);

    gemm_bf16_kernel<<<dim3(HID / 128, T_TOK / 128), 256, 0, stream>>>(
        attn_bf, wo_bf, out, T_TOK, HID, OD);
}
